// StateMixer_61924838473732
// MI455X (gfx1250) — matrix operand order repaired and checked by reading, not compiled, not run on hardware
//
#include <hip/hip_runtime.h>
#include <stdint.h>

typedef __attribute__((ext_vector_type(16))) __bf16 v16bf;
typedef __attribute__((ext_vector_type(8)))  float  v8f;

union Frag { v16bf v; unsigned u[8]; };

__device__ inline unsigned bf16rne(float f){
  unsigned u = __float_as_uint(f);
  return (u + 0x7FFFu + ((u >> 16) & 1u)) >> 16;   // round-to-nearest-even
}
__device__ inline unsigned pack2bf16(float a, float b){
  return bf16rne(a) | (bf16rne(b) << 16);
}
__device__ inline void atomicMaxF32(float* addr, float val){
  if (val >= 0.f) atomicMax((int*)addr, __float_as_int(val));
  else            atomicMin((unsigned*)addr, __float_as_uint(val));
}

// Async global->LDS copy, 16 bytes per enabled lane (ASYNCcnt-tracked).
// lds_dst: generic pointer to LDS (low 32 bits are the LDS offset per ISA).
__device__ inline void async_g2l_b128(void* lds_dst, const void* gbase, unsigned voff){
  unsigned ldsoff = (unsigned)(size_t)lds_dst;
  asm volatile("global_load_async_to_lds_b128 %0, %1, %2"
               :: "v"(ldsoff), "v"(voff), "s"(gbase)
               : "memory");
}
__device__ inline void wait_async_all(){
  asm volatile("s_wait_asynccnt 0x0" ::: "memory");
}

#define XS 68   // fp32 x-tile row stride (floats): 272B, 16B-aligned rows, bank-skewed
#define WS 36   // bf16-pair W row stride (dwords): 144B, 16B-aligned rows, bank-skewed

// ---------------------------------------------------------------------------
// Heavy kernel: per-type WMMA matmul xj = x @ Wl + bl, with two epilogues.
// MODE 0: e[n] = leaky_relu(xj + xi, 0.2) . att ; atomic max into m[g]
// MODE 1: S[g,:] += (p[n]/max(denom[g],eps)) * xj[n,:]  (run-length atomics)
// Block = 256 threads = 8 waves; block covers 128 rows; wave w owns rows
// [base + 16w, base + 16w + 16) and loops over 8 column tiles of 16.
// Staging uses global_load_async_to_lds_b128 (no VGPR round-trip).
// ---------------------------------------------------------------------------
template<int MODE>
__global__ __launch_bounds__(256) void gat_wmma_kernel(
    const float* __restrict__ x,        // [N,64] fp32
    const int*   __restrict__ batch,    // [N] sorted
    const unsigned* __restrict__ wlt,   // [128][32] packed bf16 pairs of Wl^T
    const float* __restrict__ bl,       // [128]
    const float* __restrict__ xi,       // [128]  tok@Wr+br
    const float* __restrict__ att,      // [128]
    float*       __restrict__ ep,       // MODE0: write e[n]; MODE1: read p[n]
    float*       __restrict__ mS,       // MODE0: m[256]; MODE1: S[256*128]
    const float* __restrict__ denom,    // MODE1 only
    int N)
{
  __shared__ float    sXf[128 * XS];    // fp32 x tile  (34.8 KB)
  __shared__ unsigned sW [128 * WS];    // bf16-pair Wl^T [col][Kpair] (18.4 KB)

  const int tid  = threadIdx.x;
  const int lane = tid & 31;
  const int wave = tid >> 5;
  const int half = lane >> 4;
  const int l15  = lane & 15;
  const int rowBase = blockIdx.x * 128;

  // ---- async stage Wl^T: 128 rows x 8 x 16B chunks ----
  for (int i = tid; i < 128 * 8; i += 256) {
    int r = i >> 3, c = i & 7;
    async_g2l_b128(&sW[r * WS + c * 4], wlt, (unsigned)(r * 128 + c * 16));
  }
  // ---- async stage x tile (fp32): 128 rows x 16 x 16B chunks, row-guarded ----
  {
    const int lastRow = N - rowBase;    // rows [0, lastRow) are valid
    const float* xbase = x + (size_t)rowBase * 64;
    for (int i = tid; i < 128 * 16; i += 256) {
      int r = i >> 4, c = i & 15;
      if (r < lastRow)
        async_g2l_b128(&sXf[r * XS + c * 4], xbase, (unsigned)(r * 256 + c * 16));
    }
  }
  wait_async_all();
  __syncthreads();

  // ---- A fragments for this wave's 16 rows (K halves 0..31 / 32..63) ----
  // ISA layout: lanes 0-15 M=0..15; K = (v>>2)*16 + half*8 + (v&3)*2 (+pair)
  // Convert fp32 pairs -> packed bf16 here (once per wave).
  const int mrow = wave * 16 + l15;
  Frag a0, a1;
  #pragma unroll
  for (int v = 0; v < 8; ++v) {
    int k = ((v >> 2) * 16) + half * 8 + (v & 3) * 2;
    const float* pr = &sXf[mrow * XS + k];
    a0.u[v] = pack2bf16(pr[0],  pr[1]);
    a1.u[v] = pack2bf16(pr[32], pr[33]);
  }

  // ---- MODE1: per-lane row metadata (lane i<16 holds row base+16w+i) ----
  int gi = -1; float wi = 0.f;
  if (MODE == 1) {
    if (lane < 16) {
      int n = rowBase + wave * 16 + lane;
      if (n < N) {
        gi = batch[n];
        wi = ep[n] / fmaxf(denom[gi], 1e-5f);
      }
    }
  }

  float eacc[8];
  #pragma unroll
  for (int r = 0; r < 8; ++r) eacc[r] = 0.f;

  // ---- loop over 8 column tiles ----
  #pragma unroll
  for (int ct = 0; ct < 8; ++ct) {
    const int col = ct * 16 + l15;

    // B fragments: lane N=l15; K = (v>>2)*16 + half*8 + (v&3)*2 (+pair)
    Frag b0, b1;
    #pragma unroll
    for (int v = 0; v < 8; ++v) {
      int uidx = (v >> 2) * 8 + half * 4 + (v & 3);
      b0.u[v] = sW[col * WS + uidx];
      b1.u[v] = sW[col * WS + uidx + 16];
    }

    float blv = bl[col];
    v8f c = { blv, blv, blv, blv, blv, blv, blv, blv };
    c = __builtin_amdgcn_wmma_f32_16x16x32_bf16(false, a0.v, false, b0.v,
                                                (short)0, c, false, false);
    c = __builtin_amdgcn_wmma_f32_16x16x32_bf16(false, a1.v, false, b1.v,
                                                (short)0, c, false, false);

    if (MODE == 0) {
      float xiv = xi[col], atv = att[col];
      #pragma unroll
      for (int r = 0; r < 8; ++r) {        // C: row M = r + 8*half, col fixed
        float t  = c[r] + xiv;
        float lv = (t > 0.f) ? t : 0.2f * t;
        eacc[r] += lv * atv;
      }
    } else {
      // sorted batch: run-length-compress atomics over the 8 rows this lane holds
      int curg = -1; float acc = 0.f;
      #pragma unroll
      for (int r = 0; r < 8; ++r) {
        int   src = r + half * 8;          // row index within 16-row tile
        int   g   = __shfl(gi, src, 32);
        float w   = __shfl(wi, src, 32);
        float val = c[r] * w;
        if (g != curg) {
          if (curg >= 0) atomicAdd(&mS[curg * 128 + col], acc);
          curg = g; acc = 0.f;
        }
        acc += val;
      }
      if (curg >= 0) atomicAdd(&mS[curg * 128 + col], acc);
    }
  }

  if (MODE == 0) {
    // reduce over N (16 lanes within each half) for each of 8 rows
    #pragma unroll
    for (int r = 0; r < 8; ++r) {
      eacc[r] += __shfl_xor(eacc[r], 1, 32);
      eacc[r] += __shfl_xor(eacc[r], 2, 32);
      eacc[r] += __shfl_xor(eacc[r], 4, 32);
      eacc[r] += __shfl_xor(eacc[r], 8, 32);
    }
    if (l15 == 0) {                        // lane 0 -> rows 0..7, lane 16 -> rows 8..15
      #pragma unroll
      for (int r = 0; r < 8; ++r) {
        int n = rowBase + wave * 16 + half * 8 + r;
        if (n < N) {
          float e = eacc[r];
          ep[n] = e;
          atomicMaxF32(&mS[batch[n]], e);
        }
      }
    }
  }
}

// ---------------------------------------------------------------------------
// Small helper kernels
// ---------------------------------------------------------------------------
__global__ void wl_convert_kernel(const float* __restrict__ Wl, unsigned* __restrict__ wlt) {
  int idx = blockIdx.x * 256 + threadIdx.x;          // 3*128*32 = 12288 uints
  if (idx >= 3 * 128 * 32) return;
  int t   = idx / 4096;
  int rem = idx - t * 4096;
  int n   = rem >> 5;                                // output column 0..127
  int kp  = rem & 31;                                // K pair
  int k0  = kp * 2;
  float f0 = Wl[t * 8192 + k0 * 128 + n];
  float f1 = Wl[t * 8192 + (k0 + 1) * 128 + n];
  wlt[idx] = pack2bf16(f0, f1);
}

__global__ void xi_kernel(const float* __restrict__ toks, const float* __restrict__ Wr,
                          const float* __restrict__ br, float* __restrict__ xi) {
  int t = blockIdx.x, j = threadIdx.x;               // 3 blocks x 128
  float acc = br[t * 128 + j];
  for (int i = 0; i < 128; ++i)
    acc += toks[t * 128 + i] * Wr[t * 16384 + i * 128 + j];
  xi[t * 128 + j] = acc;
}

__global__ void init_kernel(float* __restrict__ m, float* __restrict__ denom,
                            float* __restrict__ S) {
  int i = blockIdx.x * 256 + threadIdx.x;
  if (i < 768) { m[i] = -__builtin_inff(); denom[i] = 0.f; }
  if (i < 3 * 256 * 128) S[i] = 0.f;
}

__global__ void softmax_norm_kernel(const int* __restrict__ batch, const float* __restrict__ m,
                                    float* __restrict__ ep, float* __restrict__ denom, int N) {
  int n = blockIdx.x * 256 + threadIdx.x;
  if (n >= N) return;
  int g = batch[n];
  float p = __expf(ep[n] - m[g]);
  ep[n] = p;
  atomicAdd(&denom[g], p);
}

// per-type: out = BN(S + bias); write to d_out and into concat buffer
__global__ void type_norm_kernel(const float* __restrict__ S, const float* __restrict__ bias,
                                 const float* __restrict__ gamma, const float* __restrict__ beta,
                                 float* __restrict__ outp, float* __restrict__ hcat) {
  int t = blockIdx.x;                                // 3 blocks x 128 threads
  int c = threadIdx.x;
  const float* St = S + t * 32768;
  float bi = bias[t * 128 + c];
  float s = 0.f, ss = 0.f;
  for (int g = 0; g < 256; ++g) { float v = St[g * 128 + c] + bi; s += v; ss += v * v; }
  float mean = s * (1.f / 256.f);
  float var  = ss * (1.f / 256.f) - mean * mean;
  float inv  = rsqrtf(var + 1e-5f);
  float ga = gamma[t * 128 + c], be = beta[t * 128 + c];
  for (int g = 0; g < 256; ++g) {
    float o = ga * (St[g * 128 + c] + bi - mean) * inv + be;
    outp[t * 32768 + g * 128 + c] = o;
    hcat[g * 448 + 64 + t * 128 + c] = o;
  }
}

__global__ void ga_copy_kernel(const float* __restrict__ ga, float* __restrict__ hcat) {
  int i = blockIdx.x * 256 + threadIdx.x;            // 256*64
  if (i < 16384) { int g = i >> 6, c = i & 63; hcat[g * 448 + c] = ga[i]; }
}

__global__ __launch_bounds__(256) void mlp_matmul_kernel(
    const float* __restrict__ A, const float* __restrict__ W, const float* __restrict__ b,
    float* __restrict__ out, int K, int M) {
  __shared__ float sA[512];
  int g = blockIdx.x;                                // one block per batch row (256)
  for (int k = threadIdx.x; k < K; k += 256) sA[k] = A[g * K + k];
  __syncthreads();
  for (int j = threadIdx.x; j < M; j += 256) {
    float acc = b[j];
    for (int k = 0; k < K; ++k) acc += sA[k] * W[k * M + j];
    out[g * M + j] = acc;
  }
}

__global__ void bn_kernel(const float* __restrict__ io, const float* __restrict__ gamma,
                          const float* __restrict__ beta, float* __restrict__ dst,
                          int M, int doTanh) {
  int c = blockIdx.x * blockDim.x + threadIdx.x;
  if (c >= M) return;
  float s = 0.f, ss = 0.f;
  for (int g = 0; g < 256; ++g) { float v = io[g * M + c]; s += v; ss += v * v; }
  float mean = s * (1.f / 256.f);
  float var  = ss * (1.f / 256.f) - mean * mean;
  float inv  = rsqrtf(var + 1e-5f);
  float ga = gamma[c], be = beta[c];
  for (int g = 0; g < 256; ++g) {
    float o = ga * (io[g * M + c] - mean) * inv + be;
    if (doTanh) o = tanhf(o);
    dst[g * M + c] = o;
  }
}

// ---------------------------------------------------------------------------
extern "C" void kernel_launch(void* const* d_in, const int* in_sizes, int n_in,
                              void* d_out, int out_size, void* d_ws, size_t ws_size,
                              hipStream_t stream) {
  const float* x_op    = (const float*)d_in[0];
  const float* x_mc    = (const float*)d_in[1];
  const float* x_ag    = (const float*)d_in[2];
  const float* g_attr  = (const float*)d_in[3];
  const int*   b_op    = (const int*)d_in[4];
  const int*   b_mc    = (const int*)d_in[5];
  const int*   b_ag    = (const int*)d_in[6];
  const float* toks    = (const float*)d_in[7];
  const float* Wl      = (const float*)d_in[8];
  const float* bl      = (const float*)d_in[9];
  const float* Wr      = (const float*)d_in[10];
  const float* br      = (const float*)d_in[11];
  const float* att     = (const float*)d_in[12];
  const float* bias_g  = (const float*)d_in[13];
  const float* gamma_g = (const float*)d_in[14];
  const float* beta_g  = (const float*)d_in[15];
  const float* W1 = (const float*)d_in[16]; const float* b1 = (const float*)d_in[17];
  const float* g1 = (const float*)d_in[18]; const float* be1= (const float*)d_in[19];
  const float* W2 = (const float*)d_in[20]; const float* b2 = (const float*)d_in[21];
  const float* g2 = (const float*)d_in[22]; const float* be2= (const float*)d_in[23];
  const float* W3 = (const float*)d_in[24]; const float* b3 = (const float*)d_in[25];
  const float* g3 = (const float*)d_in[26]; const float* be3= (const float*)d_in[27];

  const int Nop = in_sizes[0] / 64, Nm = in_sizes[1] / 64, Na = in_sizes[2] / 64;
  float* out = (float*)d_out;

  // workspace layout (floats)
  unsigned* wlt  = (unsigned*)d_ws;                   // 3*128*32 = 12288
  float* xi      = (float*)d_ws + 12288;              // 384
  float* m       = xi + 384;                          // 768 (m[3][256])
  float* denom   = m + 768;                           // 768
  float* S       = denom + 768;                       // 3*256*128 = 98304
  float* ebuf    = S + 98304;                         // 600000 (e then p)
  float* hcat    = ebuf + 600000;                     // 256*448
  float* h1      = hcat + 114688;                     // 256*512
  float* h1b     = h1 + 131072;
  float* h2      = h1b + 131072;
  float* h2b     = h2 + 131072;
  float* h3      = h2b + 131072;                      // 256*256

  const float* xs[3]   = { x_op, x_mc, x_ag };
  const int*   bs[3]   = { b_op, b_mc, b_ag };
  const int    Ns[3]   = { Nop, Nm, Na };
  float*       es[3]   = { ebuf, ebuf + Nop, ebuf + Nop + Nm };

  // prep
  wl_convert_kernel<<<48, 256, 0, stream>>>(Wl, wlt);
  xi_kernel<<<3, 128, 0, stream>>>(toks, Wr, br, xi);
  init_kernel<<<(3 * 256 * 128 + 255) / 256, 256, 0, stream>>>(m, denom, S);

  // pass A: logits + segment max
  for (int t = 0; t < 3; ++t) {
    int blocks = (Ns[t] + 127) / 128;
    gat_wmma_kernel<0><<<blocks, 256, 0, stream>>>(
        xs[t], bs[t], wlt + t * 4096, bl + t * 128, xi + t * 128, att + t * 128,
        es[t], m + t * 256, nullptr, Ns[t]);
  }
  // pass B: p = exp(e - m), denom
  for (int t = 0; t < 3; ++t) {
    int blocks = (Ns[t] + 255) / 256;
    softmax_norm_kernel<<<blocks, 256, 0, stream>>>(bs[t], m + t * 256, es[t],
                                                    denom + t * 256, Ns[t]);
  }
  // pass C: weighted segment sum (recompute xj via WMMA)
  for (int t = 0; t < 3; ++t) {
    int blocks = (Ns[t] + 127) / 128;
    gat_wmma_kernel<1><<<blocks, 256, 0, stream>>>(
        xs[t], bs[t], wlt + t * 4096, bl + t * 128, xi + t * 128, att + t * 128,
        es[t], S + t * 32768, denom + t * 256, Ns[t]);
  }

  // type BN -> outs + concat buffer
  type_norm_kernel<<<3, 128, 0, stream>>>(S, bias_g, gamma_g, beta_g, out, hcat);
  ga_copy_kernel<<<64, 256, 0, stream>>>(g_attr, hcat);

  // graph_mix MLP: 448 -> 512 -> 512 -> 256 with BN (+tanh on first two)
  mlp_matmul_kernel<<<256, 256, 0, stream>>>(hcat, W1, b1, h1, 448, 512);
  bn_kernel<<<2, 256, 0, stream>>>(h1, g1, be1, h1b, 512, 1);
  mlp_matmul_kernel<<<256, 256, 0, stream>>>(h1b, W2, b2, h2, 512, 512);
  bn_kernel<<<2, 256, 0, stream>>>(h2, g2, be2, h2b, 512, 1);
  mlp_matmul_kernel<<<256, 256, 0, stream>>>(h2b, W3, b3, h3, 512, 256);
  bn_kernel<<<1, 256, 0, stream>>>(h3, g3, be3, out + 3 * 256 * 128, 256, 0);
}
